// GatedDeltaNet_1013612281976
// MI455X (gfx1250) — matrix operand order repaired and checked by reading, not compiled, not run on hardware
//
#include <hip/hip_runtime.h>

// ---------------- problem constants (match reference) ----------------
#define B_   2
#define T_   4096
#define D_   2048
#define H_   8
#define DK_  128
#define DV_  256
#define WC_  4
#define M_   (B_ * T_)          // 8192 rows of activations
#define NQ_  (H_ * DK_)         // 1024
#define NV_  (H_ * DV_)         // 2048
#define NCAT_ (NQ_ + NQ_ + NV_ + NV_)  // q|k|v|gate = 6144
#define CH_  32                 // staging chunk for the scan
#define EPS_NORM 1e-5f
#define EPS_L2   1e-6f

typedef __attribute__((ext_vector_type(16))) _Float16 v16h;
typedef __attribute__((ext_vector_type(8)))  float    v8f;
typedef __attribute__((ext_vector_type(4)))  unsigned int u32x4;

union HFrag { v16h v; u32x4 u[2]; };

// ---------------- f32 -> f16 convert ----------------
__global__ void k_cvt_f16(const float* __restrict__ in, _Float16* __restrict__ out, size_t n) {
  size_t i = (size_t)blockIdx.x * 256 + threadIdx.x;
  if (i < n) out[i] = (_Float16)in[i];
}

// ---------------- tiled transpose+convert: Wt[n*D_ + k] = (f16)W[k*N + n] ----------------
__global__ void __launch_bounds__(256)
k_transpose_cvt(const float* __restrict__ W, _Float16* __restrict__ Wt, int N) {
  __shared__ float tile[32][33];
  const int tx = threadIdx.x & 31;
  const int ty = threadIdx.x >> 5;            // 0..7
  const int k0 = blockIdx.y * 32;
  const int n0 = blockIdx.x * 32;
#pragma unroll
  for (int i = 0; i < 4; ++i)
    tile[ty + i * 8][tx] = W[(size_t)(k0 + ty + i * 8) * N + (n0 + tx)];
  __syncthreads();
#pragma unroll
  for (int i = 0; i < 4; ++i)
    Wt[(size_t)(n0 + ty + i * 8) * D_ + (k0 + tx)] = (_Float16)tile[tx][ty + i * 8];
}

// ---------------- WMMA GEMM: C[M,N] (f32) = A[M,K](f16 row-major) * Bt[N,K]^T ----------------
// 8 waves / block (4M x 2N wave grid); each wave owns 16(M) x 64(N): 4 WMMA tiles,
// A-fragment loaded once per K-step and reused across the 4 N-tiles.
// Prefetches are unconditional: speculative prefetch (TH=0) silently drops on
// translation failure per the CDNA5 ISA, so running past the buffer end is safe
// and keeps the K-loop a single straight-line basic block.
__global__ void __launch_bounds__(256)
k_gemm_f16_wmma(const _Float16* __restrict__ A, const _Float16* __restrict__ Bt,
                float* __restrict__ C, int M, int N, int K) {
  const int lane = threadIdx.x & 31;
  const int wave = threadIdx.x >> 5;
  const int half = lane >> 4;                  // 0: lanes 0-15, 1: lanes 16-31
  const int lr   = lane & 15;

  const int tile_m  = blockIdx.y * 64  + (wave & 3) * 16;
  const int tile_n0 = blockIdx.x * 128 + (wave >> 2) * 64;

  const _Float16* arow = A + (size_t)(tile_m + lr) * K;    // A fragment: row M = lr
  const _Float16* brow[4];
#pragma unroll
  for (int n = 0; n < 4; ++n)
    brow[n] = Bt + (size_t)(tile_n0 + n * 16 + lr) * K;    // B fragment: col N = lr

  v8f acc[4] = {};
  for (int kk = 0; kk < K; kk += 32) {
    HFrag a;
    // A 16x32 f16 layout: lanes<16 hold K in {0..7, 16..23}; lanes>=16 hold {8..15, 24..31}
    const u32x4* pa = (const u32x4*)(arow + kk + half * 8);
    a.u[0] = pa[0];          // 8 halves @ K = kk + half*8
    a.u[1] = pa[2];          // 8 halves @ K = kk + 16 + half*8
    __builtin_prefetch(arow + kk + 32 + half * 8, 0, 1);   // global_prefetch_b8

    HFrag b[4];
#pragma unroll
    for (int n = 0; n < 4; ++n) {
      // B 32x16 f16 layout: lanes<16 hold K in {0..7, 16..23}; lanes>=16 hold {8..15, 24..31}
      const u32x4* pb = (const u32x4*)(brow[n] + kk + half * 8);
      b[n].u[0] = pb[0];
      b[n].u[1] = pb[2];
      __builtin_prefetch(brow[n] + kk + 32 + half * 16, 0, 1);
    }
#pragma unroll
    for (int n = 0; n < 4; ++n)
      acc[n] = __builtin_amdgcn_wmma_f32_16x16x32_f16(false, a.v, false, b[n].v,
                                                      (short)0, acc[n], false, false);
  }
  // C layout: VGPR r -> row M = r + 8*half, col N = lr
#pragma unroll
  for (int n = 0; n < 4; ++n)
#pragma unroll
    for (int r = 0; r < 8; ++r)
      C[(size_t)(tile_m + r + half * 8) * N + tile_n0 + n * 16 + lr] = acc[n][r];
}

// ---------------- beta / g (small N=8 projections, one block per token) ----------------
__global__ void __launch_bounds__(256)
k_beta_g(const float* __restrict__ x, const float* __restrict__ Wb, const float* __restrict__ Wa,
         const float* __restrict__ A_log, const float* __restrict__ dt_bias,
         float* __restrict__ beta, float* __restrict__ g) {
  const int m = blockIdx.x;
  const int tid = threadIdx.x;
  float accb[H_] = {}, acca[H_] = {};
  for (int k = tid; k < D_; k += 256) {
    float xv = x[(size_t)m * D_ + k];
#pragma unroll
    for (int h = 0; h < H_; ++h) {
      accb[h] += xv * Wb[k * H_ + h];
      acca[h] += xv * Wa[k * H_ + h];
    }
  }
  __shared__ float red[256];
  for (int h = 0; h < H_; ++h) {
    red[tid] = accb[h]; __syncthreads();
    for (int s = 128; s > 0; s >>= 1) { if (tid < s) red[tid] += red[tid + s]; __syncthreads(); }
    float db = red[0]; __syncthreads();
    red[tid] = acca[h]; __syncthreads();
    for (int s = 128; s > 0; s >>= 1) { if (tid < s) red[tid] += red[tid + s]; __syncthreads(); }
    float da = red[0]; __syncthreads();
    if (tid == 0) {
      beta[m * H_ + h] = 1.f / (1.f + __expf(-db));
      float z = da + dt_bias[h];
      float sp = (z > 20.f) ? z : log1pf(__expf(z));
      g[m * H_ + h] = -__expf(A_log[h]) * sp;
    }
  }
}

// ---------------- causal depthwise conv (W=4) + silu + per-head l2norm ----------------
__global__ void __launch_bounds__(256)
k_conv_silu_norm(const float* __restrict__ qkvg,
                 const float* __restrict__ cwq, const float* __restrict__ cwk,
                 const float* __restrict__ cwv,
                 float* __restrict__ qn, float* __restrict__ kn, float* __restrict__ vs) {
  const int m = blockIdx.x;           // b*T + t
  const int t = m & (T_ - 1);
  const int tid = threadIdx.x;
  __shared__ float ssq[H_], ssk[H_];
  if (tid < H_) { ssq[tid] = 0.f; ssk[tid] = 0.f; }
  __syncthreads();

  float qv[4], kv[4];
#pragma unroll
  for (int j = 0; j < 4; ++j) {
    const int c = tid * 4 + j;        // channel 0..1023
    float aq = 0.f, ak = 0.f;
#pragma unroll
    for (int i = 0; i < WC_; ++i) {
      int ts = t - 3 + i;
      if (ts >= 0) {
        size_t row = (size_t)(m - 3 + i);
        aq += qkvg[row * NCAT_ + c]        * cwq[c * WC_ + i];
        ak += qkvg[row * NCAT_ + NQ_ + c]  * cwk[c * WC_ + i];
      }
    }
    aq = aq / (1.f + __expf(-aq));    // silu
    ak = ak / (1.f + __expf(-ak));
    qv[j] = aq; kv[j] = ak;
    const int h = c >> 7;             // c / DK_
    atomicAdd(&ssq[h], aq * aq);      // ds_add_f32
    atomicAdd(&ssk[h], ak * ak);
  }
  // v channels: conv + silu only
#pragma unroll
  for (int j = 0; j < 8; ++j) {
    const int c = tid * 8 + j;        // channel 0..2047
    float av = 0.f;
#pragma unroll
    for (int i = 0; i < WC_; ++i) {
      int ts = t - 3 + i;
      if (ts >= 0) av += qkvg[(size_t)(m - 3 + i) * NCAT_ + 2 * NQ_ + c] * cwv[c * WC_ + i];
    }
    vs[(size_t)m * NV_ + c] = av / (1.f + __expf(-av));
  }
  __syncthreads();
#pragma unroll
  for (int j = 0; j < 4; ++j) {
    const int c = tid * 4 + j;
    const int h = c >> 7;
    qn[(size_t)m * NQ_ + c] = qv[j] * rsqrtf(ssq[h] + EPS_L2) * 0.08838834764831845f; // DK^-0.5
    kn[(size_t)m * NQ_ + c] = kv[j] * rsqrtf(ssk[h] + EPS_L2);
  }
}

// ---------------- gated delta-rule scan ----------------
// DV columns of S evolve independently: split each (b,h) state over 4 workgroups
// (64 lanes each); every lane owns one DV column with its 128-deep K-state in VGPRs.
__global__ void __launch_bounds__(64)
k_delta_scan(const float* __restrict__ qn, const float* __restrict__ kn,
             const float* __restrict__ vs, const float* __restrict__ g,
             const float* __restrict__ beta, float* __restrict__ o) {
  __shared__ float sk[CH_ * DK_];    // 16 KB
  __shared__ float sq[CH_ * DK_];    // 16 KB
  __shared__ float sg[CH_], sb[CH_];

  const int blk = blockIdx.x;        // (b*H + h)*4 + grp
  const int grp = blk & 3;
  const int bh  = blk >> 2;
  const int b = bh / H_, h = bh % H_;
  const int tid = threadIdx.x;       // 0..63
  const int vcol = grp * 64 + tid;   // DV column this lane owns

  float S[DK_];
#pragma unroll
  for (int i = 0; i < DK_; ++i) S[i] = 0.f;

  for (int c0 = 0; c0 < T_; c0 += CH_) {
    for (int i = tid; i < CH_ * DK_; i += 64) {
      int tt = i / DK_, kk2 = i % DK_;
      size_t row = (size_t)(b * T_ + c0 + tt);
      sk[i] = kn[row * NQ_ + h * DK_ + kk2];
      sq[i] = qn[row * NQ_ + h * DK_ + kk2];
    }
    for (int i = tid; i < CH_; i += 64) {
      size_t row = (size_t)(b * T_ + c0 + i);
      sg[i] = g[row * H_ + h];
      sb[i] = beta[row * H_ + h];
    }
    __syncthreads();

    for (int tt = 0; tt < CH_; ++tt) {
      const size_t row = (size_t)(b * T_ + c0 + tt);
      const float decay = __expf(sg[tt]);
      const float bb = sb[tt];
      const float* kt = sk + tt * DK_;
      const float* qt = sq + tt * DK_;
      float err = vs[row * NV_ + h * DV_ + vcol];  // coalesced
#pragma unroll
      for (int kk2 = 0; kk2 < DK_; ++kk2) { S[kk2] *= decay; err -= kt[kk2] * S[kk2]; }
      err *= bb;
      float out = 0.f;
#pragma unroll
      for (int kk2 = 0; kk2 < DK_; ++kk2) { S[kk2] += kt[kk2] * err; out += qt[kk2] * S[kk2]; }
      o[row * NV_ + h * DV_ + vcol] = out;         // coalesced
    }
    __syncthreads();
  }
}

// ---------------- per-head gated RMSNorm -> f16 for the output GEMM ----------------
__global__ void __launch_bounds__(256)
k_gate_rmsnorm(const float* __restrict__ o, const float* __restrict__ qkvg,
               const float* __restrict__ g_norm_w, _Float16* __restrict__ og) {
  const int m = blockIdx.x;
  const int tid = threadIdx.x;       // v index
  __shared__ float red[256];
  for (int h = 0; h < H_; ++h) {
    float ov = o[(size_t)m * NV_ + h * DV_ + tid];
    red[tid] = ov * ov; __syncthreads();
    for (int s = 128; s > 0; s >>= 1) { if (tid < s) red[tid] += red[tid + s]; __syncthreads(); }
    float ms = red[0] * (1.f / DV_); __syncthreads();
    float gate = qkvg[(size_t)m * NCAT_ + 2 * NQ_ + NV_ + h * DV_ + tid];
    float sg = gate / (1.f + __expf(-gate));       // silu(gate)
    float r = ov * rsqrtf(ms + EPS_NORM) * g_norm_w[tid] * sg;
    og[(size_t)m * NV_ + h * DV_ + tid] = (_Float16)r;
  }
}

// ---------------- host-side orchestration ----------------
extern "C" void kernel_launch(void* const* d_in, const int* in_sizes, int n_in,
                              void* d_out, int out_size, void* d_ws, size_t ws_size,
                              hipStream_t stream) {
  const float* x     = (const float*)d_in[0];
  const float* Wq    = (const float*)d_in[1];
  const float* Wk    = (const float*)d_in[2];
  const float* Wv    = (const float*)d_in[3];
  const float* Wb    = (const float*)d_in[4];
  const float* Wa    = (const float*)d_in[5];
  const float* Wg    = (const float*)d_in[6];
  const float* Wo    = (const float*)d_in[7];
  const float* cwq   = (const float*)d_in[8];
  const float* cwk   = (const float*)d_in[9];
  const float* cwv   = (const float*)d_in[10];
  const float* A_log = (const float*)d_in[11];
  const float* dt_b  = (const float*)d_in[12];
  const float* gnw   = (const float*)d_in[13];
  float* out = (float*)d_out;

  char* w = (char*)d_ws;
  size_t off = 0;
  auto alloc = [&](size_t bytes) -> void* {
    void* p = w + off;
    off = (off + bytes + 255) & ~(size_t)255;
    return p;
  };
  _Float16* xbf    = (_Float16*)alloc((size_t)M_ * D_ * 2);
  _Float16* Wcat_t = (_Float16*)alloc((size_t)NCAT_ * D_ * 2);
  _Float16* Wo_t   = (_Float16*)alloc((size_t)D_ * D_ * 2);
  float*    qkvg   = (float*)   alloc((size_t)M_ * NCAT_ * 4);
  float*    qn     = (float*)   alloc((size_t)M_ * NQ_ * 4);
  float*    kn     = (float*)   alloc((size_t)M_ * NQ_ * 4);
  float*    vs     = (float*)   alloc((size_t)M_ * NV_ * 4);
  float*    betab  = (float*)   alloc((size_t)M_ * H_ * 4);
  float*    gdec   = (float*)   alloc((size_t)M_ * H_ * 4);
  float*    obuf   = (float*)   alloc((size_t)M_ * NV_ * 4);
  _Float16* og     = (_Float16*)alloc((size_t)M_ * NV_ * 2);
  (void)ws_size; (void)in_sizes; (void)n_in; (void)out_size;

  // 1) x -> f16
  {
    size_t n = (size_t)M_ * D_;
    k_cvt_f16<<<dim3((unsigned)((n + 255) / 256)), dim3(256), 0, stream>>>(x, xbf, n);
  }
  // 2) transpose+convert weights into concatenated [N,K] f16 (q|k|v|gate), and Wo
  k_transpose_cvt<<<dim3(NQ_ / 32, D_ / 32), dim3(256), 0, stream>>>(Wq, Wcat_t + (size_t)0 * D_, NQ_);
  k_transpose_cvt<<<dim3(NQ_ / 32, D_ / 32), dim3(256), 0, stream>>>(Wk, Wcat_t + (size_t)NQ_ * D_, NQ_);
  k_transpose_cvt<<<dim3(NV_ / 32, D_ / 32), dim3(256), 0, stream>>>(Wv, Wcat_t + (size_t)(2 * NQ_) * D_, NV_);
  k_transpose_cvt<<<dim3(NV_ / 32, D_ / 32), dim3(256), 0, stream>>>(Wg, Wcat_t + (size_t)(2 * NQ_ + NV_) * D_, NV_);
  k_transpose_cvt<<<dim3(D_ / 32, D_ / 32), dim3(256), 0, stream>>>(Wo, Wo_t, D_);

  // 3) fused q|k|v|gate projection GEMM (WMMA, 4 N-tiles/wave)
  k_gemm_f16_wmma<<<dim3(NCAT_ / 128, M_ / 64), dim3(256), 0, stream>>>(xbf, Wcat_t, qkvg, M_, NCAT_, D_);

  // 4) beta / g scalar projections
  k_beta_g<<<dim3(M_), dim3(256), 0, stream>>>(x, Wb, Wa, A_log, dt_b, betab, gdec);

  // 5) causal conv + silu + l2norm
  k_conv_silu_norm<<<dim3(M_), dim3(256), 0, stream>>>(qkvg, cwq, cwk, cwv, qn, kn, vs);

  // 6) gated delta-rule scan (4 column-groups per (b,h))
  k_delta_scan<<<dim3(B_ * H_ * 4), dim3(64), 0, stream>>>(qn, kn, vs, gdec, betab, obuf);

  // 7) gated RMSNorm -> f16
  k_gate_rmsnorm<<<dim3(M_), dim3(256), 0, stream>>>(obuf, qkvg, gnw, og);

  // 8) output projection GEMM (WMMA, 4 N-tiles/wave) -> d_out
  k_gemm_f16_wmma<<<dim3(D_ / 128, M_ / 64), dim3(256), 0, stream>>>(og, Wo_t, out, M_, D_, D_);
}
